// SparseAttention_12893491823331
// MI455X (gfx1250) — matrix operand order repaired and checked by reading, not compiled, not run on hardware
//
#include <hip/hip_runtime.h>
#include <hip/hip_bf16.h>

// ---------------------------------------------------------------------------
// Sparse (banded + global) attention block for MI455X (gfx1250, wave32, WMMA)
// All matmuls via v_wmma_f32_16x16x32_f16. Only local query rows computed
// (out[:, G:] is the only consumer). Working set ~120MB fits gfx1250 192MB L2.
// GEMMs use double-buffered LDS tiles fed by GLOBAL_LOAD_ASYNC_TO_LDS_B128
// (ASYNCcnt-tracked) when the toolchain exposes the builtin.
// ---------------------------------------------------------------------------

typedef _Float16 half_t;
typedef _Float16 v16h __attribute__((ext_vector_type(16)));
typedef _Float16 v8h  __attribute__((ext_vector_type(8)));
typedef float    v8f  __attribute__((ext_vector_type(8)));

constexpr int D_MODEL = 2048;
constexpr int NH      = 16;
constexpr int DH      = 128;   // head dim
constexpr int TT      = 2048;  // sequence length
constexpr int BB      = 2;     // batch
constexpr int GG      = 16;    // global tokens
constexpr int S_TOT   = GG + TT;   // 2064 rows of xg
constexpr int SP      = 2080;      // S padded to multiple of 32 (poison rows masked)
constexpr int WIN     = 256;

// --- CDNA5 async global->LDS path (probe via __has_builtin; safe fallback) ---
#if defined(__HIP_DEVICE_COMPILE__) && defined(__has_builtin)
#if __has_builtin(__builtin_amdgcn_global_load_async_to_lds_b128)
#define HAVE_ASYNC_LDS 1
#endif
#if __has_builtin(__builtin_amdgcn_s_wait_asynccnt)
#define HAVE_WAIT_ASYNC 1
#endif
#endif

#if defined(HAVE_ASYNC_LDS)
#if defined(HAVE_WAIT_ASYNC)
#define WAIT_ASYNC(n) __builtin_amdgcn_s_wait_asynccnt(n)
#else
#define WAIT_ASYNC(n) asm volatile("s_wait_asynccnt %0" ::"i"(n) : "memory")
#endif
// Builtin signature (from hipcc diagnostic): (v4i addrspace(1)*, v4i addrspace(3)*, Ii, Ii)
typedef int v4i __attribute__((vector_size(16)));
typedef __attribute__((address_space(1))) v4i* as1_v4i_p;
typedef __attribute__((address_space(3))) v4i* as3_v4i_p;
static __device__ __forceinline__ void async_cp_b128(const half_t* g, half_t* l) {
  // 16B per lane, memory -> LDS, tracked by ASYNCcnt (no VGPR data movement)
  v4i* gv = const_cast<v4i*>((const v4i*)g);
  v4i* lv = (v4i*)l;
  __builtin_amdgcn_global_load_async_to_lds_b128(
      (as1_v4i_p)gv, (as3_v4i_p)lv, /*imm offset*/ 0, /*cpol*/ 0);
}
#endif

static __device__ __forceinline__ v8f v8f_zero() {
  v8f z;
#pragma unroll
  for (int i = 0; i < 8; ++i) z[i] = 0.0f;
  return z;
}

static __device__ __forceinline__ v8f wmma_f16(v16h a, v16h b, v8f c) {
  // D = A(16x32 f16) * B(32x16 f16) + C(16x16 f32)
  return __builtin_amdgcn_wmma_f32_16x16x32_f16(false, a, false, b, (short)0, c,
                                                false, false);
}

// Build a 16-half fragment from two 8-half (16B) aligned loads.
static __device__ __forceinline__ v16h ld_frag2(const half_t* p0, const half_t* p1) {
  union { v16h v; v8h h[2]; } u;
  u.h[0] = *(const v8h*)p0;
  u.h[1] = *(const v8h*)p1;
  return u.v;
}

// ---------------------------------------------------------------------------
// Kernel 1: LayerNorm(x) -> f16 into xg rows [G..S)
// ---------------------------------------------------------------------------
__global__ __launch_bounds__(256) void ln_cast_kernel(
    const float* __restrict__ x, const float* __restrict__ sc,
    const float* __restrict__ bi, half_t* __restrict__ xg) {
  __shared__ float red[16];
  const int row  = blockIdx.x;           // 0 .. B*T-1
  const int b    = row / TT;
  const int t    = row % TT;
  const int tid  = threadIdx.x;
  const int lane = tid & 31;
  const int wid  = tid >> 5;

  const float* xr = x + (size_t)row * D_MODEL;
  float s = 0.f, ss = 0.f;
  for (int i = tid; i < D_MODEL; i += 256) {
    float v = xr[i];
    s += v; ss += v * v;
  }
#pragma unroll
  for (int d = 1; d < 32; d <<= 1) {
    s  += __shfl_xor(s, d, 32);
    ss += __shfl_xor(ss, d, 32);
  }
  if (lane == 0) { red[wid] = s; red[8 + wid] = ss; }
  __syncthreads();
  float ts = 0.f, tss = 0.f;
#pragma unroll
  for (int i = 0; i < 8; ++i) { ts += red[i]; tss += red[8 + i]; }
  const float mu   = ts / D_MODEL;
  const float var  = tss / D_MODEL - mu * mu;
  const float rstd = rsqrtf(var + 1e-5f);

  half_t* dst = xg + ((size_t)b * S_TOT + GG + t) * D_MODEL;
  for (int i = tid; i < D_MODEL; i += 256)
    dst[i] = (half_t)((xr[i] - mu) * rstd * sc[i] + bi[i]);
}

// ---------------------------------------------------------------------------
// Kernel 2: cast global tokens (raw, broadcast over batch) -> xg rows [0..G)
// ---------------------------------------------------------------------------
__global__ __launch_bounds__(256) void gcast_kernel(const float* __restrict__ g,
                                                    half_t* __restrict__ xg) {
  const int row = blockIdx.x;  // 0 .. B*G-1
  const int b = row / GG, gi = row % GG;
  const float* src = g + (size_t)gi * D_MODEL;
  half_t* dst = xg + ((size_t)b * S_TOT + gi) * D_MODEL;
  for (int i = threadIdx.x; i < D_MODEL; i += 256) dst[i] = (half_t)src[i];
}

// ---------------------------------------------------------------------------
// Kernel 3: cast a weight matrix fp32 -> f16 (row-major N x K kept as-is;
// that is exactly the natural WMMA B-fragment feed layout for y = x @ W^T)
// ---------------------------------------------------------------------------
__global__ __launch_bounds__(256) void wcast_kernel(const float* __restrict__ w,
                                                    half_t* __restrict__ wh, int n) {
  int i = blockIdx.x * 256 + threadIdx.x;
  if (i < n) wh[i] = (half_t)w[i];
}

// ---------------------------------------------------------------------------
// Kernel 4: tiled WMMA GEMM  Y[M,N] = A[M,K] * W[N,K]^T   (K = D_MODEL)
// MODE 0: store f16, head-split (b,h,s,dh), scaled by alpha  (Q / K proj)
// MODE 1: store f16, head-split transposed (b,h,dh,s)        (V proj -> V^T)
// MODE 2: store f32 + residual to d_out                      (output proj)
// Block: 256 threads = 8 waves; tile 128x128; K-step 32, double-buffered LDS
// fed by async global->LDS copies. Wave grid 4(M) x 2(N); 2x4 C-tiles/wave.
// ---------------------------------------------------------------------------
template <int MODE>
__global__ __launch_bounds__(256) void gemm_ws_kernel(
    const half_t* __restrict__ A, const half_t* __restrict__ Bw,
    void* __restrict__ dstv, const float* __restrict__ resid, int Mrows,
    float alpha) {
  __shared__ __align__(16) half_t As[2][128][40];
  __shared__ __align__(16) half_t Bs[2][128][40];

  const int tid  = threadIdx.x;
  const int lane = tid & 31;
  const int wave = tid >> 5;
  const int wm   = wave & 3;   // M group of 32 rows
  const int wn   = wave >> 2;  // N group of 64 cols
  const int n0   = blockIdx.x * 128;
  const int m0   = blockIdx.y * 128;

  v8f acc[2][4];
#pragma unroll
  for (int i = 0; i < 2; ++i)
#pragma unroll
    for (int j = 0; j < 4; ++j) acc[i][j] = v8f_zero();

  const int fm  = lane & 15;
  const int kb  = (lane >> 4) << 3;  // A-fragment K base (0 / 8)
  const int kkb = (lane >> 4) << 4;  // unused (0 / 16); B fragments use kb

  // stage one 128x32 A-tile + 128x32 B-tile into LDS buffer `buf`
  // (4 x b128 per thread; async path: ASYNCcnt += 4 per thread)
  auto stage = [&](int buf, int k0) {
#pragma unroll
    for (int it = 0; it < 2; ++it) {
      const int c   = tid + it * 256;  // 0..511
      const int row = c >> 2;
      const int col = (c & 3) * 8;
      int gm = m0 + row;
      if (gm >= Mrows) gm = Mrows - 1;  // clamp; stores predicated in epilogue
      const half_t* gA = &A[(size_t)gm * D_MODEL + k0 + col];
      const half_t* gB = &Bw[(size_t)(n0 + row) * D_MODEL + k0 + col];
#if defined(HAVE_ASYNC_LDS)
      async_cp_b128(gA, &As[buf][row][col]);
      async_cp_b128(gB, &Bs[buf][row][col]);
#else
      *(v8h*)&As[buf][row][col] = *(const v8h*)gA;
      *(v8h*)&Bs[buf][row][col] = *(const v8h*)gB;
      if (k0 + 32 < D_MODEL) {
        __builtin_prefetch(gA + 32, 0, 0);
        __builtin_prefetch(gB + 32, 0, 0);
      }
#endif
    }
  };

  constexpr int KSTEPS = D_MODEL / 32;  // 64
  stage(0, 0);
  for (int kt = 0; kt < KSTEPS; ++kt) {
    const int cur = kt & 1;
    if (kt + 1 < KSTEPS) stage(cur ^ 1, (kt + 1) * 32);
#if defined(HAVE_ASYNC_LDS)
    // async copies retire in order: <=4 outstanding means current tile landed
    if (kt + 1 < KSTEPS) { WAIT_ASYNC(4); } else { WAIT_ASYNC(0); }
#endif
    __syncthreads();

    v16h af[2], bf[4];
#pragma unroll
    for (int mi = 0; mi < 2; ++mi) {
      const half_t* p = &As[cur][wm * 32 + mi * 16 + fm][0];
      af[mi] = ld_frag2(p + kb, p + 16 + kb);
    }
#pragma unroll
    for (int ni = 0; ni < 4; ++ni) {
      const half_t* p = &Bs[cur][wn * 64 + ni * 16 + fm][0];
      bf[ni] = ld_frag2(p + kb, p + 16 + kb);
    }
#pragma unroll
    for (int mi = 0; mi < 2; ++mi)
#pragma unroll
      for (int ni = 0; ni < 4; ++ni)
        acc[mi][ni] = wmma_f16(af[mi], bf[ni], acc[mi][ni]);
    __syncthreads();
  }

  // --- epilogue: C layout is lane L -> N = L&15, row M = r + 8*(L>=16) ---
  const int rup = (lane >> 4) << 3;
#pragma unroll
  for (int mi = 0; mi < 2; ++mi)
#pragma unroll
    for (int ni = 0; ni < 4; ++ni)
#pragma unroll
      for (int r = 0; r < 8; ++r) {
        const int gm = m0 + wm * 32 + mi * 16 + r + rup;
        const int gn = n0 + wn * 64 + ni * 16 + fm;
        const float v = acc[mi][ni][r];
        if (gm < Mrows) {
          if (MODE == 0) {  // Q/K: (b,h,s,dh), padded S
            const int b = gm / S_TOT, s = gm % S_TOT;
            const int h = gn >> 7, dh = gn & 127;
            ((half_t*)dstv)[(((size_t)b * NH + h) * SP + s) * DH + dh] =
                (half_t)(v * alpha);
          } else if (MODE == 1) {  // V^T: (b,h,dh,s), padded S
            const int b = gm / S_TOT, s = gm % S_TOT;
            const int h = gn >> 7, dh = gn & 127;
            ((half_t*)dstv)[(((size_t)b * NH + h) * DH + dh) * SP + s] = (half_t)v;
          } else {  // final fp32 out + residual
            const size_t idx = (size_t)gm * D_MODEL + gn;
            ((float*)dstv)[idx] = v + resid[idx];
          }
        }
      }
}

// ---------------------------------------------------------------------------
// Kernel 5: banded+global attention, flash-style online softmax.
// Block = 128 threads = 4 independent waves; each wave owns one 16-query tile.
// Keys visited in 32-row chunks: chunk [0,32) (globals + band head), then
// chunks from floor32(q0-255) (>=32) up to q0+16. Exact per-element masking;
// padded rows (>=S_TOT) masked out. Global-query rows are skipped entirely.
// K/V fragments load straight from global (all tensors are L2-resident).
// ---------------------------------------------------------------------------
__global__ __launch_bounds__(128) void attn_kernel(
    const half_t* __restrict__ Qh, const half_t* __restrict__ Kh,
    const half_t* __restrict__ Vt, half_t* __restrict__ Oh) {
  __shared__ __align__(16) half_t Ps[4][16][40];  // per-wave P bounce (C->A layout)

  const int tid  = threadIdx.x;
  const int lane = tid & 31;
  const int w    = tid >> 5;
  const int bh   = blockIdx.x >> 5;  // T/64 = 32 query-groups per (b,h)
  const int qg   = blockIdx.x & 31;
  const int t0   = qg * 64 + w * 16;  // local token of first query in tile
  const int q0   = GG + t0;           // row in S space
  const int b    = bh >> 4;
  const int hh   = bh & 15;

  const int fm  = lane & 15;
  const int kb  = (lane >> 4) << 3;
  const int kk  = (lane >> 4) << 4;
  const int rup = (lane >> 4) << 3;

  // Q fragments (16 queries x 128 dh), Q pre-scaled by 1/sqrt(dh) at store
  const size_t qrow = ((size_t)bh * SP + q0 + fm) * DH;
  v16h qf[4];
#pragma unroll
  for (int kq = 0; kq < 4; ++kq) {
    const half_t* p = &Qh[qrow + kq * 32];
    qf[kq] = ld_frag2(p + kb, p + 16 + kb);
  }

  v8f o[8];
  float mrun[8], lrun[8];
#pragma unroll
  for (int c = 0; c < 8; ++c) o[c] = v8f_zero();
#pragma unroll
  for (int r = 0; r < 8; ++r) { mrun[r] = -1e30f; lrun[r] = 0.0f; }

  const int first = q0 - (WIN - 1);
  const int bs    = (first < 32) ? 32 : (first & ~31);
  const int limit = q0 + 16;

  int cs = 0;
  while (cs < limit) {
    // ---- scores S = Q K^T for keys [cs, cs+32) : 8 WMMAs ----
    v8f s0 = v8f_zero(), s1 = v8f_zero();
#pragma unroll
    for (int kq = 0; kq < 4; ++kq) {
      const half_t* p0 = &Kh[((size_t)bh * SP + cs + fm) * DH + kq * 32 + kb];
      const half_t* p1 = &Kh[((size_t)bh * SP + cs + 16 + fm) * DH + kq * 32 + kb];
      s0 = wmma_f16(qf[kq], ld_frag2(p0, p0 + 16), s0);
      s1 = wmma_f16(qf[kq], ld_frag2(p1, p1 + 16), s1);
    }

    // ---- mask + online softmax (row stats reduced across 16-lane halves) ----
#pragma unroll
    for (int r = 0; r < 8; ++r) {
      const int qi  = t0 + r + rup;  // local query index i
      const int k0i = cs + fm;       // absolute key row for s0 column
      const int k1i = k0i + 16;
      float v0 = s0[r], v1 = s1[r];
      const bool a0 = (k0i < S_TOT) &&
                      (k0i < GG || ((k0i - GG) <= qi && (k0i - GG) >= qi - (WIN - 1)));
      const bool a1 = (k1i < S_TOT) &&
                      (k1i < GG || ((k1i - GG) <= qi && (k1i - GG) >= qi - (WIN - 1)));
      v0 = a0 ? v0 : -1e30f;
      v1 = a1 ? v1 : -1e30f;
      float mx = fmaxf(v0, v1);
#pragma unroll
      for (int d = 1; d < 16; d <<= 1) mx = fmaxf(mx, __shfl_xor(mx, d, 16));
      const float mnew = fmaxf(mrun[r], mx);
      const float scl  = __expf(mrun[r] - mnew);
      const float p0   = __expf(v0 - mnew);
      const float p1   = __expf(v1 - mnew);
      float ps = p0 + p1;
#pragma unroll
      for (int d = 1; d < 16; d <<= 1) ps += __shfl_xor(ps, d, 16);
      lrun[r] = lrun[r] * scl + ps;
      mrun[r] = mnew;
#pragma unroll
      for (int c = 0; c < 8; ++c) o[c][r] *= scl;
      const int m = r + rup;
      Ps[w][m][fm]      = (half_t)p0;  // wave-private LDS bounce
      Ps[w][m][16 + fm] = (half_t)p1;
    }

    // ---- P as A-fragment, O += P V : 8 WMMAs ----
    const half_t* pp = &Ps[w][fm][0];
    const v16h pf = ld_frag2(pp + kb, pp + 16 + kb);
#pragma unroll
    for (int c = 0; c < 8; ++c) {
      const half_t* vp = &Vt[((size_t)bh * DH + c * 16 + fm) * SP + cs + kb];
      o[c] = wmma_f16(pf, ld_frag2(vp, vp + 16), o[c]);
    }

    cs = (cs == 0) ? bs : cs + 32;
  }

  // ---- normalize and store O as (b, tok, h*DH + dh) f16 ----
#pragma unroll
  for (int c = 0; c < 8; ++c)
#pragma unroll
    for (int r = 0; r < 8; ++r) {
      const int tok = t0 + r + rup;
      const float v = o[c][r] / lrun[r];
      Oh[((size_t)b * TT + tok) * D_MODEL + hh * DH + c * 16 + fm] = (half_t)v;
    }
}

// ---------------------------------------------------------------------------
// Host launcher
// ---------------------------------------------------------------------------
extern "C" void kernel_launch(void* const* d_in, const int* in_sizes, int n_in,
                              void* d_out, int out_size, void* d_ws,
                              size_t ws_size, hipStream_t stream) {
  const float* x   = (const float*)d_in[0];  // (B,T,D)
  const float* gt  = (const float*)d_in[1];  // (1,G,D)
  const float* lns = (const float*)d_in[2];
  const float* lnb = (const float*)d_in[3];
  const float* Wq  = (const float*)d_in[4];
  const float* Wk  = (const float*)d_in[5];
  const float* Wv  = (const float*)d_in[6];
  const float* Wo  = (const float*)d_in[7];
  float* out = (float*)d_out;

  // workspace layout (bytes, 256-aligned sections)
  constexpr size_t XG_B  = (size_t)BB * S_TOT * D_MODEL * 2;       // xg f16
  constexpr size_t W_B   = (size_t)D_MODEL * D_MODEL * 2;          // each weight f16
  constexpr size_t QKV_B = (size_t)BB * NH * SP * DH * 2;          // Q / K / V^T f16
  constexpr size_t O_B   = (size_t)BB * TT * D_MODEL * 2;          // attn out f16

  char* ws = (char*)d_ws;
  half_t* XG  = (half_t*)(ws);
  half_t* WQH = (half_t*)(ws + XG_B);
  half_t* WKH = (half_t*)(ws + XG_B + W_B);
  half_t* WVH = (half_t*)(ws + XG_B + 2 * W_B);
  half_t* WOH = (half_t*)(ws + XG_B + 3 * W_B);
  half_t* QH  = (half_t*)(ws + XG_B + 4 * W_B);
  half_t* KH  = (half_t*)(ws + XG_B + 4 * W_B + QKV_B);
  half_t* VT  = (half_t*)(ws + XG_B + 4 * W_B + 2 * QKV_B);
  half_t* OH  = (half_t*)(ws + XG_B + 4 * W_B + 3 * QKV_B);
  (void)ws_size; (void)in_sizes; (void)n_in; (void)out_size; (void)O_B;

  // 1) LayerNorm + f16 cast, and raw global tokens into xg
  ln_cast_kernel<<<BB * TT, 256, 0, stream>>>(x, lns, lnb, XG);
  gcast_kernel<<<BB * GG, 256, 0, stream>>>(gt, XG);

  // 2) weights -> f16
  const int wn = D_MODEL * D_MODEL;
  const int wg = (wn + 255) / 256;
  wcast_kernel<<<wg, 256, 0, stream>>>(Wq, WQH, wn);
  wcast_kernel<<<wg, 256, 0, stream>>>(Wk, WKH, wn);
  wcast_kernel<<<wg, 256, 0, stream>>>(Wv, WVH, wn);
  wcast_kernel<<<wg, 256, 0, stream>>>(Wo, WOH, wn);

  // 3) projections (M = B*(G+T) = 4128 -> 33 row-blocks with edge predication)
  const int Mp = BB * S_TOT;
  dim3 gp(D_MODEL / 128, (Mp + 127) / 128);
  const float qscale = 0.08838834764831845f;  // 1/sqrt(128)
  gemm_ws_kernel<0><<<gp, 256, 0, stream>>>(XG, WQH, QH, nullptr, Mp, qscale);
  gemm_ws_kernel<0><<<gp, 256, 0, stream>>>(XG, WKH, KH, nullptr, Mp, 1.0f);
  gemm_ws_kernel<1><<<gp, 256, 0, stream>>>(XG, WVH, VT, nullptr, Mp, 1.0f);

  // 4) attention over local query tiles only
  attn_kernel<<<BB * NH * (TT / 64), 128, 0, stream>>>(QH, KH, VT, OH);

  // 5) output projection + residual -> fp32 d_out
  const int Mo = BB * TT;
  dim3 go(D_MODEL / 128, Mo / 128);
  gemm_ws_kernel<2><<<go, 256, 0, stream>>>(OH, WOH, out, x, Mo, 1.0f);
}
